// LinearAttention_28561532518577
// MI455X (gfx1250) — compile-verified, matrix operand order checked by reading, not run on hardware
//
#include <hip/hip_runtime.h>
#include <hip/hip_bf16.h>

typedef __attribute__((ext_vector_type(16))) __bf16 v16bf;
typedef __attribute__((ext_vector_type(8)))  __bf16 v8bf;    // 16 bytes
typedef __attribute__((ext_vector_type(8)))  float  v8f;

// ---------------------------------------------------------------------------
// f32 -> bf16 elementwise convert
__global__ void la_cvt_f32_bf16(const float* __restrict__ in,
                                __bf16* __restrict__ out, long long n) {
    long long i = (long long)blockIdx.x * blockDim.x + threadIdx.x;
    long long stride = (long long)gridDim.x * blockDim.x;
    for (; i < n; i += stride) out[i] = (__bf16)in[i];
}

// f32 [R,C] -> bf16 transposed [C,R]:  out[c*R + r] = in[r*C + c]
__global__ void la_transpose_f32_bf16(const float* __restrict__ in,
                                      __bf16* __restrict__ out, int R, int C) {
    long long n = (long long)R * C;
    long long i = (long long)blockIdx.x * blockDim.x + threadIdx.x;
    long long stride = (long long)gridDim.x * blockDim.x;
    for (; i < n; i += stride) {
        int r = (int)(i / C);
        int c = (int)(i % C);
        out[(long long)c * R + r] = (__bf16)in[i];
    }
}

// ---------------------------------------------------------------------------
// Batched bf16 GEMM, both operands k-contiguous:
//   C[b0,b1] = alpha * A[b0,b1] (MxK, row stride lda) @ Bt[b0,b1]^T + bias
// where Bt is stored [Ncol][K] (row stride ldb, K contiguous).
// Each wave32 computes a 64x32 tile = 4x2 v_wmma_f32_16x16x32_bf16 fragments
// (12 global_load_b128 per 8 WMMAs). C store is generic-strided, f32 or bf16.
__device__ __forceinline__ v16bf la_load_frag(const __bf16* rowp, int k, int half) {
    v8bf lo = *(const v8bf*)(rowp + k + half * 8);
    v8bf hi = *(const v8bf*)(rowp + k + 16 + half * 8);
    v16bf f;
    #pragma unroll
    for (int i = 0; i < 8; ++i) { f[i] = lo[i]; f[i + 8] = hi[i]; }
    return f;
}

__global__ void la_wmma_gemm_bf16(
    const __bf16* __restrict__ A, const __bf16* __restrict__ Bt,
    void* __restrict__ C, const float* __restrict__ bias,
    long long a0, long long a1, long long lda,
    long long b0, long long b1, long long ldb,
    long long c0, long long c1, long long csm, long long csn,
    int nb0, int nb1, int M, int Ncol, int K,
    float alpha, int store_bf16)
{
    const int lane = threadIdx.x & 31;
    const int l15  = lane & 15;
    const int half = lane >> 4;          // 0: K {0..7,16..23}; 1: K {8..15,24..31}
    const int wid  = threadIdx.x >> 5;

    const long long waves_total = (long long)gridDim.x * (blockDim.x >> 5);
    long long wave = (long long)blockIdx.x * (blockDim.x >> 5) + wid;

    const int mtiles = M >> 6;           // 64-row wave tiles
    const int ntiles = Ncol >> 5;        // 32-col wave tiles
    const long long tiles = (long long)nb0 * nb1 * mtiles * ntiles;

    for (long long t = wave; t < tiles; t += waves_total) {
        int nt = (int)(t % ntiles);
        long long r = t / ntiles;
        int mt = (int)(r % mtiles);
        r /= mtiles;
        int bb1 = (int)(r % nb1);
        int bb0 = (int)(r / nb1);

        // per-lane row pointers (k-contiguous rows)
        const __bf16* Ar[4];
        Ar[0] = A + bb0 * a0 + bb1 * a1 + (long long)(mt * 64 + l15) * lda;
        #pragma unroll
        for (int ms = 1; ms < 4; ++ms) Ar[ms] = Ar[ms - 1] + 16 * lda;

        const __bf16* Br[2];
        Br[0] = Bt + bb0 * b0 + bb1 * b1 + (long long)(nt * 32 + l15) * ldb;
        Br[1] = Br[0] + 16 * ldb;

        v8f acc[4][2];
        #pragma unroll
        for (int ms = 0; ms < 4; ++ms)
            #pragma unroll
            for (int ns = 0; ns < 2; ++ns)
                acc[ms][ns] = (v8f){0.f,0.f,0.f,0.f,0.f,0.f,0.f,0.f};

        for (int k = 0; k < K; k += 32) {
            #pragma unroll
            for (int ms = 0; ms < 4; ++ms) __builtin_prefetch(Ar[ms] + k + 128, 0, 1);
            #pragma unroll
            for (int ns = 0; ns < 2; ++ns) __builtin_prefetch(Br[ns] + k + 128, 0, 1);

            v16bf fb[2];
            #pragma unroll
            for (int ns = 0; ns < 2; ++ns) fb[ns] = la_load_frag(Br[ns], k, half);

            #pragma unroll
            for (int ms = 0; ms < 4; ++ms) {
                v16bf fa = la_load_frag(Ar[ms], k, half);
                acc[ms][0] = __builtin_amdgcn_wmma_f32_16x16x32_bf16(
                                 false, fa, false, fb[0], (short)0, acc[ms][0], false, false);
                acc[ms][1] = __builtin_amdgcn_wmma_f32_16x16x32_bf16(
                                 false, fa, false, fb[1], (short)0, acc[ms][1], false, false);
            }
        }

        // Store 4x2 fragments. C/D layout: VGPR i -> row = base + half*8 + i, col = l15.
        const long long cbase = bb0 * c0 + bb1 * c1;
        #pragma unroll
        for (int ms = 0; ms < 4; ++ms) {
            #pragma unroll
            for (int ns = 0; ns < 2; ++ns) {
                int row0 = mt * 64 + ms * 16 + half * 8;
                int col  = nt * 32 + ns * 16 + l15;
                float bv = bias ? bias[col] : 0.f;
                long long off = cbase + (long long)row0 * csm + (long long)col * csn;
                #pragma unroll
                for (int i = 0; i < 8; ++i) {
                    float v = alpha * acc[ms][ns][i] + bv;
                    if (store_bf16) ((__bf16*)C)[off + (long long)i * csm] = (__bf16)v;
                    else            ((float*)C)[off + (long long)i * csm] = v;
                }
            }
        }
    }
}

// ---------------------------------------------------------------------------
extern "C" void kernel_launch(void* const* d_in, const int* in_sizes, int n_in,
                              void* d_out, int out_size, void* d_ws, size_t ws_size,
                              hipStream_t stream) {
    (void)in_sizes; (void)n_in; (void)out_size; (void)ws_size;

    constexpr int B = 4, N = 2048, DIM = 512, H = 8, HD = 64;
    constexpr long long TOK = (long long)B * N;            // 8192 tokens
    const float scale_over_n = (1.0f / 8.0f) / (float)N;   // 1/sqrt(HD)/N

    const float* x     = (const float*)d_in[0];  // [B,N,DIM]
    const float* w_qkv = (const float*)d_in[1];  // [DIM, 3*H*HD]
    const float* w_out = (const float*)d_in[2];  // [H*HD, DIM]
    const float* b_out = (const float*)d_in[3];  // [DIM]
    float* out = (float*)d_out;                  // [B,N,DIM] f32

    // Workspace (all bf16, all offsets multiples of 8 elements)
    __bf16* x_bf    = (__bf16*)d_ws;                        // [TOK, DIM]
    __bf16* wqkv_t  = x_bf   + TOK * DIM;                   // [1536, 512]  (n,k)
    __bf16* wout_t  = wqkv_t + (long long)3 * DIM * DIM;    // [512, 512]   (n,k)
    __bf16* Qb      = wout_t + (long long)DIM * DIM;        // [B,H,N,HD]
    __bf16* Kt      = Qb     + TOK * DIM;                   // [B,H,HD,N]
    __bf16* Vt      = Kt     + TOK * DIM;                   // [B,H,HD,N]
    __bf16* KVt     = Vt     + TOK * DIM;                   // [B,H,HD,HD] ((K^T V)^T)
    __bf16* Ob      = KVt    + (long long)B * H * HD * HD;  // [TOK, H*HD]

    const int BLK = 256;                 // 8 waves
    const int WPB = BLK / 32;

    // ---- Pass 0: precision / layout conversion ----
    la_cvt_f32_bf16<<<2048, BLK, 0, stream>>>(x, x_bf, TOK * DIM);
    la_transpose_f32_bf16<<<1024, BLK, 0, stream>>>(w_qkv, wqkv_t, DIM, 3 * DIM);
    la_transpose_f32_bf16<<<512,  BLK, 0, stream>>>(w_out, wout_t, DIM, DIM);

    const long long bNHD  = (long long)H * N * HD;   // per-b stride of Q/Kt/Vt (=1048576)
    const long long hNHD  = (long long)N * HD;       // per-h stride            (=131072)

    // ---- Stage 1: Q / K^T / V^T projections, batched over (b,h) ----
    {
        int M = N, Nc = HD, K = DIM;
        long long tiles = (long long)B * H * (M / 64) * (Nc / 32);
        int blocks = (int)((tiles + WPB - 1) / WPB);
        // Q: [B,H,N,HD] normal
        la_wmma_gemm_bf16<<<blocks, BLK, 0, stream>>>(
            x_bf, wqkv_t, Qb, nullptr,
            (long long)N * DIM, 0LL, DIM,
            0LL, (long long)HD * DIM, DIM,
            bNHD, hNHD, HD, 1LL,
            B, H, M, Nc, K, 1.0f, 1);
        // K^T: [B,H,HD,N] (csm=1, csn=N)
        la_wmma_gemm_bf16<<<blocks, BLK, 0, stream>>>(
            x_bf, wqkv_t + (long long)DIM * DIM, Kt, nullptr,
            (long long)N * DIM, 0LL, DIM,
            0LL, (long long)HD * DIM, DIM,
            bNHD, hNHD, 1LL, (long long)N,
            B, H, M, Nc, K, 1.0f, 1);
        // V^T: [B,H,HD,N]
        la_wmma_gemm_bf16<<<blocks, BLK, 0, stream>>>(
            x_bf, wqkv_t + (long long)2 * DIM * DIM, Vt, nullptr,
            (long long)N * DIM, 0LL, DIM,
            0LL, (long long)HD * DIM, DIM,
            bNHD, hNHD, 1LL, (long long)N,
            B, H, M, Nc, K, 1.0f, 1);
    }

    // ---- Stage 2: KV^T[b,h] = (K^T V)^T, stored [e][d] ----
    {
        int M = HD, Nc = HD, K = N;
        long long tiles = (long long)B * H * (M / 64) * (Nc / 32);
        int blocks = (int)((tiles + WPB - 1) / WPB);
        la_wmma_gemm_bf16<<<blocks, BLK, 0, stream>>>(
            Kt, Vt, KVt, nullptr,
            bNHD, hNHD, (long long)N,
            bNHD, hNHD, (long long)N,
            (long long)H * HD * HD, (long long)HD * HD, 1LL, (long long)HD,
            B, H, M, Nc, K, 1.0f, 1);
    }

    // ---- Stage 3: O[b,h] = (scale/N) * Q[b,h] @ KV[b,h] -> O[tok, h*HD+e] ----
    {
        int M = N, Nc = HD, K = HD;
        long long tiles = (long long)B * H * (M / 64) * (Nc / 32);
        int blocks = (int)((tiles + WPB - 1) / WPB);
        la_wmma_gemm_bf16<<<blocks, BLK, 0, stream>>>(
            Qb, KVt, Ob, nullptr,
            bNHD, hNHD, (long long)HD,
            (long long)H * HD * HD, (long long)HD * HD, (long long)HD,
            (long long)N * (H * HD), (long long)HD, (long long)(H * HD), 1LL,
            B, H, M, Nc, K, scale_over_n, 1);
    }

    // ---- Stage 4: out[tok,512] = O @ w_out + b_out (f32 store) ----
    {
        int M = (int)TOK, Nc = DIM, K = H * HD;
        long long tiles = (long long)(M / 64) * (Nc / 32);
        int blocks = (int)((tiles + WPB - 1) / WPB);
        la_wmma_gemm_bf16<<<blocks, BLK, 0, stream>>>(
            Ob, wout_t, out, b_out,
            0LL, 0LL, (long long)(H * HD),
            0LL, 0LL, (long long)DIM,
            0LL, 0LL, (long long)DIM, 1LL,
            1, 1, M, Nc, K, 1.0f, 0);
    }
}
